// ContinuousSpatialMambaBlock_V3_71330816852042
// MI455X (gfx1250) — matrix operand order repaired and checked by reading, not compiled, not run on hardware
//
#include <hip/hip_runtime.h>
#include <hip/hip_bf16.h>
#include <math.h>

// ---------------------------------------------------------------------------
// Types for CDNA5 WMMA / async copy
// ---------------------------------------------------------------------------
typedef __attribute__((ext_vector_type(16))) __bf16          bf16x16;
typedef __attribute__((ext_vector_type(8)))  float           floatx8;
typedef __attribute__((ext_vector_type(16))) unsigned short  ushort16;
typedef __attribute__((ext_vector_type(4)))  int             i32x4;

#define GLOBAL_AS __attribute__((address_space(1)))
#define LDS_AS    __attribute__((address_space(3)))

#define DIM_D   512
#define DIM_DI  1024
#define DIM_B   8
#define DIM_N   4096              // 64 x 64
#define DIM_M   (DIM_B * DIM_N)   // 32768 GEMM rows
#define HW_H    64
#define HW_W    64

// gfx1250 async global->LDS copy path (guarded; falls back to load+store)
#if defined(__has_builtin)
#if __has_builtin(__builtin_amdgcn_global_load_async_to_lds_b128) && \
    __has_builtin(__builtin_amdgcn_s_wait_asynccnt)
#define USE_ASYNC_COPY 1
#endif
#endif

__device__ __forceinline__ unsigned short f32_to_bf16(float f) {
    unsigned int x = __builtin_bit_cast(unsigned int, f);
    unsigned int r = (x + 0x7FFFu + ((x >> 16) & 1u)) >> 16;
    return (unsigned short)r;
}
__device__ __forceinline__ float bf16_to_f32(unsigned short h) {
    unsigned int x = ((unsigned int)h) << 16;
    return __builtin_bit_cast(float, x);
}
__device__ __forceinline__ float silu_f(float v) {
    return v / (1.0f + __expf(-v));
}

// ---------------------------------------------------------------------------
// fp32 (K x N, row-major) -> bf16 transposed (N x K, row-major).
// ---------------------------------------------------------------------------
__global__ __launch_bounds__(256)
void transpose_cvt_bf16_kernel(const float* __restrict__ src,
                               unsigned short* __restrict__ dst,
                               int K, int N) {
    __shared__ float tile[32][33];
    const int kb = blockIdx.y * 32;
    const int nb = blockIdx.x * 32;
    const int tx = threadIdx.x & 31;
    const int ty = threadIdx.x >> 5;   // 0..7
    #pragma unroll
    for (int r = 0; r < 32; r += 8)
        tile[r + ty][tx] = src[(size_t)(kb + r + ty) * N + nb + tx];
    __syncthreads();
    #pragma unroll
    for (int r = 0; r < 32; r += 8)
        dst[(size_t)(nb + r + ty) * K + kb + tx] =
            f32_to_bf16(tile[tx][r + ty]);
}

// ---------------------------------------------------------------------------
// LayerNorm over D=512, output bf16 (GEMM A operand)
// ---------------------------------------------------------------------------
__global__ __launch_bounds__(256)
void layernorm_bf16_kernel(const float* __restrict__ x,
                           const float* __restrict__ gamma,
                           const float* __restrict__ beta,
                           unsigned short* __restrict__ xn) {
    const size_t row = blockIdx.x;
    const int tid = threadIdx.x;
    const float* xr = x + row * DIM_D;
    float2 v = *reinterpret_cast<const float2*>(xr + tid * 2);
    float s  = v.x + v.y;
    float s2 = v.x * v.x + v.y * v.y;
    #pragma unroll
    for (int o = 16; o > 0; o >>= 1) {
        s  += __shfl_xor(s,  o);
        s2 += __shfl_xor(s2, o);
    }
    __shared__ float ws[8], ws2[8];
    const int wave = tid >> 5, lane = tid & 31;
    if (lane == 0) { ws[wave] = s; ws2[wave] = s2; }
    __syncthreads();
    if (wave == 0) {
        float a  = (lane < 8) ? ws[lane]  : 0.0f;
        float a2 = (lane < 8) ? ws2[lane] : 0.0f;
        #pragma unroll
        for (int o = 4; o > 0; o >>= 1) {
            a  += __shfl_xor(a,  o);
            a2 += __shfl_xor(a2, o);
        }
        if (lane == 0) { ws[0] = a; ws2[0] = a2; }
    }
    __syncthreads();
    const float mean = ws[0] * (1.0f / DIM_D);
    const float var  = ws2[0] * (1.0f / DIM_D) - mean * mean;
    const float rstd = rsqrtf(var + 1e-5f);
    const float y0 = (v.x - mean) * rstd * gamma[tid * 2 + 0] + beta[tid * 2 + 0];
    const float y1 = (v.y - mean) * rstd * gamma[tid * 2 + 1] + beta[tid * 2 + 1];
    unsigned int packed = (unsigned int)f32_to_bf16(y0) |
                          ((unsigned int)f32_to_bf16(y1) << 16);
    *reinterpret_cast<unsigned int*>(xn + row * DIM_D + tid * 2) = packed;
}

// ---------------------------------------------------------------------------
// Depthwise 3x3 conv (zero pad) + bias + SiLU, channel-last (B,N,Di).
// ---------------------------------------------------------------------------
__global__ __launch_bounds__(256)
void conv_local_silu_kernel(const float* __restrict__ up,
                            const float* __restrict__ w,     // (Di,1,3,3)
                            const float* __restrict__ bias,  // (Di)
                            float* __restrict__ u,
                            unsigned short* __restrict__ ubf) {
    const int p = blockIdx.x;
    const int bimg = p >> 12;
    const int sp   = p & 4095;
    const int y = sp >> 6, x = sp & 63;
    const size_t base = (size_t)bimg * DIM_N;
    for (int c = threadIdx.x; c < DIM_DI; c += 256) {
        float acc = bias[c];
        const float* wc = w + c * 9;
        #pragma unroll
        for (int dy = -1; dy <= 1; ++dy) {
            const int yy = y + dy;
            if (yy < 0 || yy >= HW_H) continue;
            #pragma unroll
            for (int dx = -1; dx <= 1; ++dx) {
                const int xx = x + dx;
                if (xx < 0 || xx >= HW_W) continue;
                acc += up[(base + yy * HW_W + xx) * DIM_DI + c] *
                       wc[(dy + 1) * 3 + (dx + 1)];
            }
        }
        const float s = silu_f(acc);
        const size_t o = (base + sp) * DIM_DI + c;
        u[o]   = s;
        ubf[o] = f32_to_bf16(s);
    }
}

// ---------------------------------------------------------------------------
// One reaction-diffusion Euler step (replicate-pad 3x3 depthwise conv).
// ---------------------------------------------------------------------------
__global__ __launch_bounds__(256)
void euler_step_kernel(const float* __restrict__ h,
                       const float* __restrict__ delta,
                       const float* __restrict__ wdiff,   // (Di,1,3,3)
                       const float* __restrict__ alpha,   // (Di)
                       const float* __restrict__ betar,   // (Di)
                       float* __restrict__ hn,
                       unsigned short* __restrict__ hbf,  // nullable
                       float dtv) {
    const int p = blockIdx.x;
    const int bimg = p >> 12;
    const int sp   = p & 4095;
    const int y = sp >> 6, x = sp & 63;
    const size_t base = (size_t)bimg * DIM_N;
    for (int c = threadIdx.x; c < DIM_DI; c += 256) {
        const float* wc = wdiff + c * 9;
        float diff = 0.0f;
        #pragma unroll
        for (int dy = -1; dy <= 1; ++dy) {
            const int yy = min(max(y + dy, 0), HW_H - 1);
            #pragma unroll
            for (int dx = -1; dx <= 1; ++dx) {
                const int xx = min(max(x + dx, 0), HW_W - 1);
                diff += h[(base + yy * HW_W + xx) * DIM_DI + c] *
                        wc[(dy + 1) * 3 + (dx + 1)];
            }
        }
        const size_t o = (base + sp) * DIM_DI + c;
        const float hc = h[o];
        const float f  = delta[o] * diff + alpha[c] * hc - betar[c] * hc * hc * hc;
        const float v  = hc + dtv * f;
        hn[o] = v;
        if (hbf) hbf[o] = f32_to_bf16(v);
    }
}

// ---------------------------------------------------------------------------
// bf16 WMMA GEMM:  C(MxN) = A(MxK) * Bt(NxK)^T.
// Block tile 128x128, K-stage 64 (two WMMA K-steps per barrier round),
// double-buffered LDS fed by global_load_async_to_lds_b128; ASYNCcnt-gated
// pipeline overlaps the DMA for stage i+1 with WMMA compute of stage i.
// ---------------------------------------------------------------------------
#define BK      64
#define LDS_K   72   // ushorts per row (64 data + 8 pad); 144B rows, 16B-aligned chunks

__device__ __forceinline__ void copy16(const unsigned short* src,
                                       unsigned short* dst) {
#if defined(USE_ASYNC_COPY)
    __builtin_amdgcn_global_load_async_to_lds_b128(
        (GLOBAL_AS i32x4*)(void*)src,
        (LDS_AS i32x4*)(void*)dst, 0, 0);
#else
    uint4 v = *reinterpret_cast<const uint4*>(src);
    *reinterpret_cast<uint4*>(dst) = v;
#endif
}

// stage a 128 x 64 bf16 tile (row-major, ld elements) into LDS buffer
__device__ __forceinline__ void stage_tile(const unsigned short* g, int ld,
                                           int row0, int k0,
                                           unsigned short* lds, int tid) {
    #pragma unroll
    for (int t = 0; t < 4; ++t) {
        const int idx = tid + t * 256;
        const int r = idx >> 3;      // 0..127
        const int s = idx & 7;       // 8-bf16 segment
        copy16(g + (size_t)(row0 + r) * ld + k0 + s * 8,
               &lds[r * LDS_K + s * 8]);
    }
}

__device__ __forceinline__ bf16x16 load_a_frag(const unsigned short* As,
                                               int tile_m, int koff, int lane) {
    const int m = tile_m + (lane & 15);
    const int g = lane >> 4;
    const unsigned short* row = As + m * LDS_K + koff;
    ushort16 a;
    #pragma unroll
    for (int j = 0; j < 8; ++j) {
        const int k = (j < 4) ? (g * 8 + 2 * j) : (16 + g * 8 + 2 * (j - 4));
        unsigned int pair = *reinterpret_cast<const unsigned int*>(row + k);
        a[2 * j + 0] = (unsigned short)(pair & 0xFFFFu);
        a[2 * j + 1] = (unsigned short)(pair >> 16);
    }
    return __builtin_bit_cast(bf16x16, a);
}

__device__ __forceinline__ bf16x16 load_b_frag(const unsigned short* Bs,
                                               int tile_n, int koff, int lane) {
    const int n = tile_n + (lane & 15);
    const int g = lane >> 4;
    const unsigned short* row = Bs + n * LDS_K + koff;
    ushort16 b;
    #pragma unroll
    for (int j = 0; j < 8; ++j) {
        const int k = (j < 4) ? (g * 8 + 2 * j) : (16 + g * 8 + 2 * (j - 4));
        unsigned int pair = *reinterpret_cast<const unsigned int*>(row + k);
        b[2 * j + 0] = (unsigned short)(pair & 0xFFFFu);
        b[2 * j + 1] = (unsigned short)(pair >> 16);
    }
    return __builtin_bit_cast(bf16x16, b);
}

template <int EPI>
__global__ __launch_bounds__(256, 2)
void gemm_bf16_kernel(const unsigned short* __restrict__ A, int lda,
                      const unsigned short* __restrict__ Bt, int ldb,
                      int K, int ldc,
                      float* __restrict__ outF,
                      unsigned short* __restrict__ outH,
                      const float* __restrict__ bias,        // EPI2
                      const float* __restrict__ u,           // EPI3
                      const float* __restrict__ Dp,          // EPI3
                      const unsigned short* __restrict__ sz, // EPI3
                      const float* __restrict__ resid) {     // EPI4
    __shared__ unsigned short As[2][128 * LDS_K];
    __shared__ unsigned short Bs[2][128 * LDS_K];

    const int tid  = threadIdx.x;
    const int wave = tid >> 5, lane = tid & 31;
    const int wm = wave & 3;          // 4 row-groups of 32
    const int wn = wave >> 2;         // 2 col-groups of 64
    const int m0 = blockIdx.y * 128;
    const int n0 = blockIdx.x * 128;

    floatx8 acc[2][4] = {};

    const int nstage = K / BK;
    // prologue: stage 0 into buffer 0
    stage_tile(A,  lda, m0, 0, As[0], tid);
    stage_tile(Bt, ldb, n0, 0, Bs[0], tid);

    for (int it = 0; it < nstage; ++it) {
        const int cur = it & 1;
        const bool has_next = (it + 1 < nstage);
        if (has_next) {
            const int k1 = (it + 1) * BK;
            stage_tile(A,  lda, m0, k1, As[cur ^ 1], tid);
            stage_tile(Bt, ldb, n0, k1, Bs[cur ^ 1], tid);
        }
#if defined(USE_ASYNC_COPY)
        // async loads complete in order: <=8 outstanding means stage `it`'s
        // 8 copies have landed; stage `it+1`'s DMA stays in flight under WMMA
        if (has_next) __builtin_amdgcn_s_wait_asynccnt(8);
        else          __builtin_amdgcn_s_wait_asynccnt(0);
#endif
        __syncthreads();

        #pragma unroll
        for (int kk = 0; kk < 2; ++kk) {
            const int koff = kk * 32;
            bf16x16 af[2];
            #pragma unroll
            for (int i = 0; i < 2; ++i)
                af[i] = load_a_frag(As[cur], wm * 32 + i * 16, koff, lane);
            bf16x16 bf[4];
            #pragma unroll
            for (int j = 0; j < 4; ++j)
                bf[j] = load_b_frag(Bs[cur], wn * 64 + j * 16, koff, lane);

            #pragma unroll
            for (int i = 0; i < 2; ++i)
                #pragma unroll
                for (int j = 0; j < 4; ++j)
                    acc[i][j] = __builtin_amdgcn_wmma_f32_16x16x32_bf16(
                        false, af[i], false, bf[j], (short)0, acc[i][j],
                        false, false);
        }
        __syncthreads();   // all waves done reading buf[cur] before rewrite
    }

    // ---- epilogue: C tile -> row = r + 8*(lane/16), col = lane%16 ----------
    const int lrow = (lane >> 4) * 8;
    const int lcol = lane & 15;
    #pragma unroll
    for (int i = 0; i < 2; ++i) {
        #pragma unroll
        for (int j = 0; j < 4; ++j) {
            const int gmb = m0 + wm * 32 + i * 16 + lrow;
            const int gn  = n0 + wn * 64 + j * 16 + lcol;
            #pragma unroll
            for (int r = 0; r < 8; ++r) {
                const size_t idx = (size_t)(gmb + r) * ldc + gn;
                const float v = acc[i][j][r];
                if constexpr (EPI == 0) {
                    outF[idx] = v;
                } else if constexpr (EPI == 1) {
                    outH[idx] = f32_to_bf16(silu_f(v));
                } else if constexpr (EPI == 2) {
                    const float t = v + bias[gn];
                    const float sp = (t > 20.0f) ? t : log1pf(__expf(t));
                    outF[idx] = fminf(sp, 0.15f);
                } else if constexpr (EPI == 3) {
                    const float y = v + u[idx] * Dp[gn];
                    outH[idx] = f32_to_bf16(y * bf16_to_f32(sz[idx]));
                } else { // EPI == 4
                    outF[idx] = resid[idx] + v;
                }
            }
        }
    }
}

// ---------------------------------------------------------------------------
// Host-side launch
// ---------------------------------------------------------------------------
extern "C" void kernel_launch(void* const* d_in, const int* in_sizes, int n_in,
                              void* d_out, int out_size, void* d_ws, size_t ws_size,
                              hipStream_t stream) {
    (void)in_sizes; (void)n_in; (void)out_size; (void)ws_size;
    const float* x        = (const float*)d_in[0];
    const float* ln_gamma = (const float*)d_in[1];
    const float* ln_beta  = (const float*)d_in[2];
    const float* W_in     = (const float*)d_in[3];
    const float* conv_lw  = (const float*)d_in[4];
    const float* conv_lb  = (const float*)d_in[5];
    const float* W_dt     = (const float*)d_in[6];
    const float* b_dt     = (const float*)d_in[7];
    const float* D_param  = (const float*)d_in[8];
    const float* conv_dw  = (const float*)d_in[9];
    const float* alpha    = (const float*)d_in[10];
    const float* beta_r   = (const float*)d_in[11];
    const float* W_ssm    = (const float*)d_in[12];
    const float* W_out    = (const float*)d_in[13];
    float* out = (float*)d_out;

    // workspace carve-up
    char* ws = (char*)d_ws;
    size_t off = 0;
    auto alloc = [&](size_t bytes) {
        size_t o = off; off += (bytes + 255) & ~(size_t)255; return o;
    };
    const size_t M = DIM_M;
    unsigned short* xn_bf  = (unsigned short*)(ws + alloc(M * DIM_D * 2));
    unsigned short* win_t  = (unsigned short*)(ws + alloc((size_t)2 * DIM_DI * DIM_D * 2)); // (2Di) x D
    unsigned short* wdt_t  = (unsigned short*)(ws + alloc((size_t)DIM_DI * DIM_DI * 2));    // Di x Di
    unsigned short* wssm_t = (unsigned short*)(ws + alloc((size_t)DIM_DI * DIM_DI * 2));
    unsigned short* wout_t = (unsigned short*)(ws + alloc((size_t)DIM_D * DIM_DI * 2));     // D x Di
    float*          u_pre  = (float*)(ws + alloc(M * DIM_DI * 4));
    float*          u_f32  = (float*)(ws + alloc(M * DIM_DI * 4));
    unsigned short* u_bf   = (unsigned short*)(ws + alloc(M * DIM_DI * 2));
    unsigned short* sz_bf  = (unsigned short*)(ws + alloc(M * DIM_DI * 2));
    float*          delta  = (float*)(ws + alloc(M * DIM_DI * 4));
    float*          h0     = (float*)(ws + alloc(M * DIM_DI * 4));
    float*          h1     = (float*)(ws + alloc(M * DIM_DI * 4));
    unsigned short* h_bf   = (unsigned short*)(ws + alloc(M * DIM_DI * 2));
    unsigned short* g_bf   = (unsigned short*)(ws + alloc(M * DIM_DI * 2));

    // 1) weights: fp32 (KxN) -> bf16 transposed (NxK)
    transpose_cvt_bf16_kernel<<<dim3(2 * DIM_DI / 32, DIM_D / 32), 256, 0, stream>>>(
        W_in, win_t, DIM_D, 2 * DIM_DI);
    transpose_cvt_bf16_kernel<<<dim3(DIM_DI / 32, DIM_DI / 32), 256, 0, stream>>>(
        W_dt, wdt_t, DIM_DI, DIM_DI);
    transpose_cvt_bf16_kernel<<<dim3(DIM_DI / 32, DIM_DI / 32), 256, 0, stream>>>(
        W_ssm, wssm_t, DIM_DI, DIM_DI);
    transpose_cvt_bf16_kernel<<<dim3(DIM_D / 32, DIM_DI / 32), 256, 0, stream>>>(
        W_out, wout_t, DIM_DI, DIM_D);

    // 2) LayerNorm -> bf16
    layernorm_bf16_kernel<<<M, 256, 0, stream>>>(x, ln_gamma, ln_beta, xn_bf);

    const dim3 blk(256);
    // 3) u_pre = xn @ W_in[:, :Di]   (Bt rows 0..Di-1)
    gemm_bf16_kernel<0><<<dim3(DIM_DI / 128, M / 128), blk, 0, stream>>>(
        xn_bf, DIM_D, win_t, DIM_D, DIM_D, DIM_DI,
        u_pre, nullptr, nullptr, nullptr, nullptr, nullptr, nullptr);
    // 4) sz = silu(xn @ W_in[:, Di:])  (Bt rows Di..2Di-1)
    gemm_bf16_kernel<1><<<dim3(DIM_DI / 128, M / 128), blk, 0, stream>>>(
        xn_bf, DIM_D, win_t + (size_t)DIM_DI * DIM_D, DIM_D, DIM_D, DIM_DI,
        nullptr, sz_bf, nullptr, nullptr, nullptr, nullptr, nullptr);

    // 5) depthwise 3x3 + bias + SiLU -> u (fp32 + bf16)
    conv_local_silu_kernel<<<M, 256, 0, stream>>>(u_pre, conv_lw, conv_lb,
                                                  u_f32, u_bf);

    // 6) delta = min(softplus(u @ W_dt + b_dt), 0.15)
    gemm_bf16_kernel<2><<<dim3(DIM_DI / 128, M / 128), blk, 0, stream>>>(
        u_bf, DIM_DI, wdt_t, DIM_DI, DIM_DI, DIM_DI,
        delta, nullptr, b_dt, nullptr, nullptr, nullptr, nullptr);

    // 7) 4 Euler steps (dt = 0.25); final step also emits bf16
    const float dtv = 0.25f;
    euler_step_kernel<<<M, 256, 0, stream>>>(u_f32, delta, conv_dw, alpha,
                                             beta_r, h0, nullptr, dtv);
    euler_step_kernel<<<M, 256, 0, stream>>>(h0, delta, conv_dw, alpha,
                                             beta_r, h1, nullptr, dtv);
    euler_step_kernel<<<M, 256, 0, stream>>>(h1, delta, conv_dw, alpha,
                                             beta_r, h0, nullptr, dtv);
    euler_step_kernel<<<M, 256, 0, stream>>>(h0, delta, conv_dw, alpha,
                                             beta_r, h1, h_bf, dtv);

    // 8) g = (h @ W_ssm_out + u*D_param) * silu(z)
    gemm_bf16_kernel<3><<<dim3(DIM_DI / 128, M / 128), blk, 0, stream>>>(
        h_bf, DIM_DI, wssm_t, DIM_DI, DIM_DI, DIM_DI,
        nullptr, g_bf, nullptr, u_f32, D_param, sz_bf, nullptr);

    // 9) out = residual + g @ W_out
    gemm_bf16_kernel<4><<<dim3(DIM_D / 128, M / 128), blk, 0, stream>>>(
        g_bf, DIM_DI, wout_t, DIM_DI, DIM_DI, DIM_D,
        out, nullptr, nullptr, nullptr, nullptr, nullptr, x);
}
